// Transfusion_455266533743
// MI455X (gfx1250) — matrix operand order repaired and checked by reading, not compiled, not run on hardware
//
#include <hip/hip_runtime.h>

typedef __attribute__((ext_vector_type(16))) _Float16 v16h;
typedef __attribute__((ext_vector_type(8)))  _Float16 v8h;
typedef __attribute__((ext_vector_type(4)))  _Float16 v4h;
typedef __attribute__((ext_vector_type(8)))  float    v8f;

#define TB 4
#define TH 16
#define TS 2048
#define TD 64
#define TM 4

#define WAVES 4
#define QTILE 64     // queries per block (16 per wave, WMMA N dimension)
#define JT    64     // keys per iteration
#define KP    72     // halves: pitch of lds_k [key][d] / lds_vt [d][key]; 144B rows -> 16B-aligned frags
#define OPIT  68     // floats: pitch of output staging

__device__ __forceinline__ float fast_exp2(float x) {
    return __builtin_amdgcn_exp2f(x);   // v_exp_f32
}

// Two 16B fragment reads p[0..7], p[16..23] (16B-aligned): used for both K and V^T A-fragments.
// Thanks to the K-index permutation trick, A-fragments are two contiguous 8-half runs in LDS.
__device__ __forceinline__ v16h lds_frag32(const _Float16* p) {
    union { v16h v; v8h h[2]; } u;
    u.h[0] = *(const v8h*)p;
    u.h[1] = *(const v8h*)(p + 16);
    return u.v;
}

__global__ __launch_bounds__(128)
void transfusion_flash_wmma3(const float* __restrict__ qg,
                             const float* __restrict__ kg,
                             const float* __restrict__ vg,
                             const int*   __restrict__ mods,
                             float*       __restrict__ og)
{
    __shared__ _Float16 lds_k [JT * KP];   // [key][d]   f16
    __shared__ _Float16 lds_vt[TD * KP];   // [d][key]   f16 (transposed V)
    __shared__ float    lds_o [QTILE * OPIT];

    const int tid  = threadIdx.x;
    const int wave = tid >> 5;
    const int lane = tid & 31;
    const int l16  = lane & 15;
    const int hi   = lane >> 4;

    const int nqt = TS / QTILE;
    const int qt  = blockIdx.x % nqt;
    const int bh  = blockIdx.x / nqt;
    const int b   = bh / TH;

    const size_t head = (size_t)bh * TS * TD;
    const float* Q = qg + head;
    const float* K = kg + head;
    const float* V = vg + head;
    float*       O = og + head;

    const int q0 = qt * QTILE;

    // ---- modality spans ----
    int offs[TM], ends[TM];
#pragma unroll
    for (int m = 0; m < TM; ++m) {
        const int off = mods[(b * TM + m) * 3 + 1];
        const int len = mods[(b * TM + m) * 3 + 2];
        offs[m] = off; ends[m] = off + len;
    }
    // tight j upper bound for this block (allowed-key set is a prefix per row)
    const int i_max = q0 + QTILE - 1;
    int j_hi = i_max + 1;
#pragma unroll
    for (int m = 0; m < TM; ++m)
        if (offs[m] <= i_max && ends[m] > j_hi) j_hi = ends[m];
    const int n_jt = (j_hi + JT - 1) / JT;

    // ---- per-lane query index and mask prefix limit ----
    const int iq = q0 + wave * 16 + l16;          // this lane's query (N dim)
    int jlim = iq;                                // allow j <= jlim
#pragma unroll
    for (int m = 0; m < TM; ++m)
        if (iq >= offs[m] && ends[m] - 1 > jlim) jlim = ends[m] - 1;

    // ---- Q B-fragments, packed for the permuted d-index mapping ----
    // WMMA K-index -> d mapping: K{0..7}->d{hi? 8..15 : 0..7} etc. B-layout element e:
    //   lane<16 : e<8 -> d = kc*32 + e        ; e>=8 -> d = kc*32 + 16 + (e-8)
    //   lane>=16: e<8 -> d = kc*32 + 8 + e    ; e>=8 -> d = kc*32 + 24 + (e-8)
    // (A side then reads two 8-half runs: lds_k[key][kc*32 + hi*8 + {0..7, 16..23}])
    const float qscale = 0.125f * 1.44269504088896340736f;   // D^-0.5 * log2(e)
    const float* qrow = Q + (size_t)iq * TD;
    v16h qb[2];
#pragma unroll
    for (int kc = 0; kc < 2; ++kc) {
        const int dbase = kc * 32 + hi * 8;
#pragma unroll
        for (int e = 0; e < 8; ++e) {
            qb[kc][e]     = (_Float16)(qrow[dbase + e]      * qscale);
            qb[kc][8 + e] = (_Float16)(qrow[dbase + 16 + e] * qscale);
        }
    }

    // ---- accumulators: O^T fragments (M=d, N=query), per-lane softmax state ----
    v8f acc[4];
#pragma unroll
    for (int dt = 0; dt < 4; ++dt)
#pragma unroll
        for (int r = 0; r < 8; ++r) acc[dt][r] = 0.0f;
    float mrun = -3.0e38f, lrun = 0.0f;

    for (int jt = 0; jt < n_jt; ++jt) {
        const int j0 = jt * JT;

        __syncthreads();   // previous tile fully consumed

        if (jt + 1 < n_jt) {   // prefetch next K/V tile (one 128B line per thread)
            const size_t nb = (size_t)(j0 + JT) * TD + tid * 32;
            __builtin_prefetch(K + nb, 0, 1);
            __builtin_prefetch(V + nb, 0, 1);
        }

        // ---- K tile: coalesced b128 loads, packed f16 row-major store ----
#pragma unroll
        for (int e = 0; e < 8; ++e) {
            const int idx = (tid + 128 * e) * 4;
            const int r = idx >> 6, c = idx & 63;
            const float4 kq = *(const float4*)(K + (size_t)(j0 + r) * TD + c);
            v4h kh;
            kh[0] = (_Float16)kq.x; kh[1] = (_Float16)kq.y;
            kh[2] = (_Float16)kq.z; kh[3] = (_Float16)kq.w;
            *(v4h*)&lds_k[r * KP + c] = kh;
        }
        // ---- V tile: per-d coalesced b32 gathers -> packed b64 transposed store ----
        {
            const int vd  = tid & 63;
            const int vkg = (tid >> 6) * 4;
#pragma unroll
            for (int p2 = 0; p2 < 8; ++p2) {
                const int kb = p2 * 8 + vkg;
                const float* vcol = V + (size_t)(j0 + kb) * TD + vd;
                const float x0 = vcol[0 * TD], x1 = vcol[1 * TD];
                const float x2 = vcol[2 * TD], x3 = vcol[3 * TD];
                v4h vh;
                vh[0] = (_Float16)x0; vh[1] = (_Float16)x1;
                vh[2] = (_Float16)x2; vh[3] = (_Float16)x3;
                *(v4h*)&lds_vt[vd * KP + kb] = vh;
            }
        }
        __syncthreads();

        // ---- S^T = K * Q^T : 4 fragments (M=16 keys each, N=16 queries) ----
        v8f sf[4];
#pragma unroll
        for (int nt = 0; nt < 4; ++nt) {
            v8f cf;
#pragma unroll
            for (int r = 0; r < 8; ++r) cf[r] = 0.0f;
#pragma unroll
            for (int kc = 0; kc < 2; ++kc) {
                // A-fragment d = kc*32 + hi*8 + {0..7, 16..23} (permuted d-index)
                const v16h ka = lds_frag32(&lds_k[(nt * 16 + l16) * KP + kc * 32 + hi * 8]);
                cf = __builtin_amdgcn_wmma_f32_16x16x32_f16(
                        false, ka, false, qb[kc], (short)0, cf, false, false);
            }
            sf[nt] = cf;
        }

        // ---- mask (skipped entirely for interior tiles; uniform branch) ----
        bool full = (j0 + JT - 1 <= q0);
#pragma unroll
        for (int m = 0; m < TM; ++m)
            full = full || (offs[m] <= q0 && ends[m] >= j0 + JT);
        if (!full) {
#pragma unroll
            for (int nt = 0; nt < 4; ++nt)
#pragma unroll
                for (int r = 0; r < 8; ++r) {
                    const int j = j0 + nt * 16 + hi * 8 + r;
                    if (j > jlim) sf[nt][r] = -1.0e30f;
                }
        }

        // ---- online softmax: per-lane scalar stats, one half-wave shuffle each ----
        float x = -3.0e38f;
#pragma unroll
        for (int nt = 0; nt < 4; ++nt)
#pragma unroll
            for (int r = 0; r < 8; ++r) x = fmaxf(x, sf[nt][r]);
        x = fmaxf(x, __shfl_xor(x, 16, 32));
        const float mnew = fmaxf(mrun, x);
        const float fac  = fast_exp2(mrun - mnew);
        mrun = mnew;
        float sum = 0.0f;
#pragma unroll
        for (int nt = 0; nt < 4; ++nt)
#pragma unroll
            for (int r = 0; r < 8; ++r) {
                const float pv = fast_exp2(sf[nt][r] - mnew);
                sf[nt][r] = pv;
                sum += pv;
            }
        sum += __shfl_xor(sum, 16, 32);
        lrun = lrun * fac + sum;
#pragma unroll
        for (int dt = 0; dt < 4; ++dt)
#pragma unroll
            for (int r = 0; r < 8; ++r) acc[dt][r] *= fac;

        // ---- O^T += V^T * P^T ----
        // Permuted key-index mapping K{0..7,8..15,16..23,24..31} ->
        // keys kc*32 + {hi*8+0..7, 16+hi*8+0..7, ...}: the P^T B-fragment is just the
        // lane's own sf values (no cross-lane ops), and the V^T A-fragment is a
        // two 16B reads at lds_vt[d][kc*32 + hi*8 + {0..7, 16..23}].
#pragma unroll
        for (int kc = 0; kc < 2; ++kc) {
            v16h pb;
#pragma unroll
            for (int e = 0; e < 8; ++e) {
                pb[e]     = (_Float16)sf[2 * kc][e];       // key kc*32 + hi*8 + e
                pb[8 + e] = (_Float16)sf[2 * kc + 1][e];   // key kc*32 + 16 + hi*8 + e
            }
#pragma unroll
            for (int dt = 0; dt < 4; ++dt) {
                const v16h va = lds_frag32(&lds_vt[(dt * 16 + l16) * KP + kc * 32 + hi * 8]);
                acc[dt] = __builtin_amdgcn_wmma_f32_16x16x32_f16(
                             false, va, false, pb, (short)0, acc[dt], false, false);
            }
        }
    }

    // ---- epilogue: normalize, transpose through LDS, coalesced b128 stores ----
    __syncthreads();
    const float invl = 1.0f / lrun;
#pragma unroll
    for (int dt = 0; dt < 4; ++dt)
#pragma unroll
        for (int r = 0; r < 8; ++r)
            lds_o[(wave * 16 + l16) * OPIT + dt * 16 + hi * 8 + r] = acc[dt][r] * invl;
    __syncthreads();
#pragma unroll
    for (int e = 0; e < 8; ++e) {
        const int idx = (tid + 128 * e) * 4;
        const int row = idx >> 6, col = idx & 63;
        *(float4*)(O + (size_t)(q0 + row) * TD + col) = *(const float4*)&lds_o[row * OPIT + col];
    }
}

extern "C" void kernel_launch(void* const* d_in, const int* in_sizes, int n_in,
                              void* d_out, int out_size, void* d_ws, size_t ws_size,
                              hipStream_t stream) {
    (void)in_sizes; (void)n_in; (void)out_size; (void)d_ws; (void)ws_size;
    const float* q    = (const float*)d_in[0];
    const float* k    = (const float*)d_in[1];
    const float* v    = (const float*)d_in[2];
    const int*   mods = (const int*)d_in[3];
    float* out = (float*)d_out;

    const int blocks = TB * TH * (TS / QTILE);   // 2048
    transfusion_flash_wmma3<<<blocks, WAVES * 32, 0, stream>>>(q, k, v, mods, out);
}
